// VectorQuantizer_90263032693002
// MI455X (gfx1250) — matrix operand order repaired and checked by reading, not compiled, not run on hardware
//
#include <hip/hip_runtime.h>
#include <hip/hip_bf16.h>

// ---------------------------------------------------------------------------
// VQ-VAE vector quantizer for MI455X (gfx1250, wave32, WMMA + async-to-LDS).
//   inputs : (16, 262144) f32   -> N=65536 vectors, D=64
//   emb_w  : (4096, 64)   f32   -> K=4096 codebook
// Outputs (flat f32): loss(1) | quantized_st(4194304) | perplexity(1)
//                   | encoding_indices(65536) | encodings one-hot(268435456)
// argmin(dist) == argmax(x.e - 0.5*||e||^2): fused bf16 WMMA GEMM + running max.
// Codebook chunks staged global->LDS via GLOBAL_LOAD_ASYNC_TO_LDS (ASYNCcnt).
// ---------------------------------------------------------------------------

#define VQ_D 64
#define VQ_K 4096
#define VQ_N 65536            // B*T
#define VQ_CHUNK 1024         // codebook entries staged in LDS per pass
#define VQ_T 4096
#define VQ_B 16

#define Q_OFF     1
#define Q_SIZE    4194304
#define PERP_OFF  4194305
#define IDXF_OFF  4194306
#define ENC_OFF   4259842
#define ENC_SIZE  268435456LL

typedef __attribute__((ext_vector_type(16))) __bf16 v16bf;
typedef __attribute__((ext_vector_type(8)))  float  v8f;
typedef __attribute__((ext_vector_type(4)))  float  fvec4;
typedef __attribute__((ext_vector_type(4)))  unsigned int uvec4;

// -------------------------- accumulator init -------------------------------
__global__ void vq_init_kernel(float* accs) {
  accs[0] = 0.0f;   // loss accumulator
  accs[1] = 0.0f;   // perplexity entropy accumulator
}

// ---------------- codebook half-norms + bf16 conversion --------------------
__global__ void __launch_bounds__(256) vq_norms_kernel(
    const float* __restrict__ CB, float* __restrict__ HN,
    __bf16* __restrict__ CBb) {
  int k = blockIdx.x * 256 + threadIdx.x;           // k < 4096
  const float* r = CB + (size_t)k * VQ_D;
  float acc = 0.0f;
  #pragma unroll 8
  for (int d = 0; d < VQ_D; ++d) {
    float v = r[d];
    acc += v * v;
    CBb[(size_t)k * VQ_D + d] = (__bf16)v;
  }
  HN[k] = 0.5f * acc;
}

// ------------------- fused WMMA GEMM + argmax kernel -----------------------
// Block = 256 threads = 8 waves; each wave owns 16 input rows.
// LDS: VQ_CHUNK codebook rows as bf16 (128KB) + their half-norms (4KB),
// staged with async global->LDS copies (no VGPR round trip).
__global__ void __launch_bounds__(256) vq_argmin_kernel(
    const float* __restrict__ X, const __bf16* __restrict__ CBb,
    const float* __restrict__ HN, int* __restrict__ IDX) {
  extern __shared__ __align__(128) char smem_raw[];
  __bf16* lds_cb = (__bf16*)smem_raw;                        // CHUNK*64 bf16
  float*  lds_hn = (float*)(smem_raw + VQ_CHUNK * VQ_D * 2); // CHUNK f32

  // wave-relative LDS byte offsets (shared aperture keeps offset in addr[31:0])
  const unsigned int lds_cb_off = (unsigned int)(uintptr_t)lds_cb;
  const unsigned int lds_hn_off = (unsigned int)(uintptr_t)lds_hn;

  const int tid  = threadIdx.x;
  const int lane = tid & 31;
  const int wave = tid >> 5;
  const int half = lane >> 4;         // 0: lanes 0-15, 1: lanes 16-31
  const int nloc = lane & 15;
  const int rows_base = blockIdx.x * 128 + wave * 16;

  // --- A fragments (16x64 inputs tile, two K=32 bf16 fragments), loop-invariant.
  // A layout chosen here (B matches): lane holds row m=lane&15; 16 contiguous K from half*16.
  const float* xr = X + (size_t)(rows_base + nloc) * VQ_D;
  __builtin_prefetch(xr, 0, 1);                     // global_prefetch_b8
  const int kb = half * 16;
  v16bf a0, a1;
  #pragma unroll
  for (int j = 0; j < 8; ++j) {
    a0[j]     = (__bf16)xr[kb + j];
    a0[8 + j] = (__bf16)xr[8 + kb + j];
    a1[j]     = (__bf16)xr[32 + kb + j];
    a1[8 + j] = (__bf16)xr[40 + kb + j];
  }

  float bestv[8];
  int   besti[8];
  #pragma unroll
  for (int v = 0; v < 8; ++v) { bestv[v] = -3.0e38f; besti[v] = 0; }

  const int kb2 = half * 16;          // B layout: 16 contiguous K per half-wave

  for (int cb = 0; cb < VQ_K; cb += VQ_CHUNK) {
    __syncthreads();                  // previous chunk's LDS reads complete
    { // async stage: 128KB codebook chunk + 4KB half-norms, global -> LDS
      const char* gsrc = (const char*)(CBb + (size_t)cb * VQ_D);
      #pragma unroll 4
      for (int i = tid; i < VQ_CHUNK * VQ_D * 2 / 16; i += 256) {
        unsigned long long ga = (unsigned long long)(uintptr_t)(gsrc + (size_t)i * 16);
        unsigned int       la = lds_cb_off + (unsigned int)i * 16u;
        asm volatile("global_load_async_to_lds_b128 %0, %1, off"
                     :: "v"(la), "v"(ga) : "memory");
      }
      const char* hsrc = (const char*)(HN + cb);
      for (int i = tid; i < VQ_CHUNK; i += 256) {
        unsigned long long ga = (unsigned long long)(uintptr_t)(hsrc + (size_t)i * 4);
        unsigned int       la = lds_hn_off + (unsigned int)i * 4u;
        asm volatile("global_load_async_to_lds_b32 %0, %1, off"
                     :: "v"(la), "v"(ga) : "memory");
      }
      asm volatile("s_wait_asynccnt 0x0" ::: "memory");
    }
    __syncthreads();

    #pragma unroll 1
    for (int t = 0; t < VQ_CHUNK / 16; ++t) {
      const int eloc = t * 16 + nloc;         // this lane's codebook column
      // B fragments: lane = column n, 16 contiguous K values -> 32B LDS loads
      const v16bf b0 = *(const v16bf*)(lds_cb + eloc * VQ_D + kb2);
      const v16bf b1 = *(const v16bf*)(lds_cb + eloc * VQ_D + 32 + kb2);
      v8f c = {0.f, 0.f, 0.f, 0.f, 0.f, 0.f, 0.f, 0.f};
      c = __builtin_amdgcn_wmma_f32_16x16x32_bf16(false, a0, false, b0,
                                                  (short)0, c, false, false);
      c = __builtin_amdgcn_wmma_f32_16x16x32_bf16(false, a1, false, b1,
                                                  (short)0, c, false, false);
      const float hn = lds_hn[eloc];
      const int e = cb + eloc;
      #pragma unroll
      for (int v = 0; v < 8; ++v) {           // c[v]: row = v + half*8, col = nloc
        float s = c[v] - hn;
        if (s > bestv[v]) { bestv[v] = s; besti[v] = e; }
      }
    }
  }

  // reduce (score, index) across the 16 lanes sharing each output row
  #pragma unroll
  for (int off = 1; off < 16; off <<= 1) {
    #pragma unroll
    for (int v = 0; v < 8; ++v) {
      float ov = __shfl_xor(bestv[v], off, 32);
      int   oi = __shfl_xor(besti[v], off, 32);
      if (ov > bestv[v] || (ov == bestv[v] && oi < besti[v])) {
        bestv[v] = ov; besti[v] = oi;
      }
    }
  }
  if (nloc == 0) {
    #pragma unroll
    for (int v = 0; v < 8; ++v)
      IDX[rows_base + half * 8 + v] = besti[v];
  }
}

// --------------- gather quantized + loss reduction -------------------------
__global__ void __launch_bounds__(256) vq_quant_kernel(
    const float* __restrict__ X, const float* __restrict__ CB,
    const int* __restrict__ IDX, float* __restrict__ QOUT,
    float* __restrict__ loss_acc) {
  __shared__ float wsum[8];
  int tid = blockIdx.x * 256 + threadIdx.x;   // tid < N*D = 4194304
  int n = tid >> 6, d = tid & 63;
  int k = IDX[n];
  float q = CB[(size_t)k * VQ_D + d];
  float x = X[tid];
  QOUT[tid] = q;                 // straight-through value == quantized
  float diff = q - x;
  float s = diff * diff;
  #pragma unroll
  for (int off = 16; off > 0; off >>= 1) s += __shfl_xor(s, off, 32);
  if ((threadIdx.x & 31) == 0) wsum[threadIdx.x >> 5] = s;
  __syncthreads();
  if (threadIdx.x == 0) {
    float t = 0.0f;
    #pragma unroll
    for (int w = 0; w < 8; ++w) t += wsum[w];
    atomicAdd(loss_acc, t);
  }
}

// --------------------- one-hot zero fill (NT B128) -------------------------
__global__ void __launch_bounds__(256) vq_zero4_kernel(fvec4* __restrict__ p,
                                                       long long n4) {
  long long i = (long long)blockIdx.x * 256 + threadIdx.x;
  const long long stride = (long long)gridDim.x * 256;
  fvec4 z = {0.f, 0.f, 0.f, 0.f};
  for (; i < n4; i += stride) __builtin_nontemporal_store(z, p + i);
}

__global__ void vq_zero_tails_kernel(float* head, int nh, float* tail, int nt) {
  int i = threadIdx.x;
  if (i < nh) head[i] = 0.0f;
  if (i < nt) tail[i] = 0.0f;
}

// --------------------- scatter ones + float indices ------------------------
__global__ void __launch_bounds__(256) vq_scatter_kernel(
    const int* __restrict__ IDX, float* __restrict__ ENC,
    float* __restrict__ IDXF) {
  int n = blockIdx.x * 256 + threadIdx.x;     // n < 65536
  int k = IDX[n];
  ENC[(size_t)n * VQ_K + k] = 1.0f;
  IDXF[n] = (float)k;
}

// ------------------------- perplexity entropy ------------------------------
// avg_probs over batch dim (B=16): per t, sum_b (1/16)*log(c_b/16 + 1e-5)
// where c_b = multiplicity of idx[b,t] among the 16 batch entries.
__global__ void __launch_bounds__(256) vq_perp_kernel(
    const int* __restrict__ IDX, float* __restrict__ perp_acc) {
  __shared__ float wsum[8];
  int t = blockIdx.x * 256 + threadIdx.x;     // t < 4096
  int my[VQ_B];
  #pragma unroll
  for (int b = 0; b < VQ_B; ++b) my[b] = IDX[b * VQ_T + t];
  float s = 0.0f;
  #pragma unroll
  for (int b = 0; b < VQ_B; ++b) {
    int c = 0;
    #pragma unroll
    for (int b2 = 0; b2 < VQ_B; ++b2) c += (my[b2] == my[b]) ? 1 : 0;
    s += __logf((float)c * 0.0625f + 1e-5f);
  }
  s *= 0.0625f;
  #pragma unroll
  for (int off = 16; off > 0; off >>= 1) s += __shfl_xor(s, off, 32);
  if ((threadIdx.x & 31) == 0) wsum[threadIdx.x >> 5] = s;
  __syncthreads();
  if (threadIdx.x == 0) {
    float tot = 0.0f;
    #pragma unroll
    for (int w = 0; w < 8; ++w) tot += wsum[w];
    atomicAdd(perp_acc, tot);
  }
}

// ------------------------------ finalize -----------------------------------
__global__ void vq_final_kernel(const float* __restrict__ accs,
                                float* __restrict__ out) {
  // loss = q_latent + 0.25*e_latent; both equal mean((q-x)^2)
  out[0] = 1.25f * accs[0] * (1.0f / 4194304.0f);
  out[PERP_OFF] = __expf(-accs[1]);
}

// ---------------------------------------------------------------------------
extern "C" void kernel_launch(void* const* d_in, const int* in_sizes, int n_in,
                              void* d_out, int out_size, void* d_ws, size_t ws_size,
                              hipStream_t stream) {
  const float* X  = (const float*)d_in[0];    // (16, 262144)
  const float* CB = (const float*)d_in[1];    // (4096, 64)
  float* out = (float*)d_out;

  char* ws = (char*)d_ws;
  float*  HN  = (float*)ws;                                // 16 KB
  __bf16* CBb = (__bf16*)(ws + 16384);                     // 512 KB
  int*    IDX = (int*)(ws + 16384 + 524288);               // 256 KB
  float*  ACC = (float*)(ws + 16384 + 524288 + 262144);    // 2 floats

  vq_init_kernel<<<1, 1, 0, stream>>>(ACC);
  vq_norms_kernel<<<VQ_K / 256, 256, 0, stream>>>(CB, HN, CBb);

  const size_t lds_bytes = VQ_CHUNK * VQ_D * 2 + VQ_CHUNK * 4;  // 132 KB
  vq_argmin_kernel<<<VQ_N / 128, 256, lds_bytes, stream>>>(X, CBb, HN, IDX);

  vq_quant_kernel<<<(VQ_N * VQ_D) / 256, 256, 0, stream>>>(X, CB, IDX,
                                                           out + Q_OFF, ACC);

  // zero the 1.07 GB one-hot region with 16B-aligned NT stores
  float* ENC = out + ENC_OFF;
  uintptr_t a = (uintptr_t)ENC;
  int head = (int)(((16 - (a & 15)) & 15) >> 2);
  long long n4 = (ENC_SIZE - head) / 4;
  int tail = (int)(ENC_SIZE - head - n4 * 4);
  vq_zero_tails_kernel<<<1, 32, 0, stream>>>(ENC, head, ENC + head + n4 * 4, tail);
  vq_zero4_kernel<<<4096, 256, 0, stream>>>((fvec4*)(ENC + head), n4);

  vq_scatter_kernel<<<VQ_N / 256, 256, 0, stream>>>(IDX, ENC, out + IDXF_OFF);
  vq_perp_kernel<<<VQ_T / 256, 256, 0, stream>>>(IDX, ACC + 1);
  vq_final_kernel<<<1, 1, 0, stream>>>(ACC, out);
}
